// GptOssExperts_11424613007749
// MI455X (gfx1250) — compile-verified, matrix operand order checked by reading, not run on hardware
//
#include <hip/hip_runtime.h>

#define ACT_ALPHA 1.702f
#define ACT_LIMIT 7.0f

typedef __bf16 bf16_t;
typedef __bf16 v2bf  __attribute__((ext_vector_type(2)));
typedef __bf16 v4bf  __attribute__((ext_vector_type(4)));
typedef __bf16 v8bf  __attribute__((ext_vector_type(8)));
typedef __bf16 v16bf __attribute__((ext_vector_type(16)));
typedef float  v4f   __attribute__((ext_vector_type(4)));
typedef float  v8f   __attribute__((ext_vector_type(8)));

static constexpr int T_TOK = 2048;     // B*S tokens
static constexpr int HDIM  = 2880;     // hidden
static constexpr int DDIM  = 2880;     // intermediate
static constexpr int NEXP  = 8;
static constexpr int GUDIM = 2 * DDIM; // 5760

// ---------------------------------------------------------------------------
// CDNA5 async global->LDS copy (ASYNCcnt-tracked, bypasses VGPRs).
// Per-lane: 32 bytes (two b128 copies; IOFFSET applies to both LDS and
// global address, so offset:16 covers the second half).
// ---------------------------------------------------------------------------
__device__ inline void async_copy32(bf16_t* lds_dst, const bf16_t* gsrc) {
    unsigned lds = (unsigned)(uintptr_t)lds_dst;   // addr[31:0] = LDS offset
    asm volatile(
        "global_load_async_to_lds_b128 %0, %1, off\n\t"
        "global_load_async_to_lds_b128 %0, %1, off offset:16"
        :: "v"(lds), "v"(gsrc)
        : "memory");
}

__device__ inline void wait_async0() {
#if __has_builtin(__builtin_amdgcn_s_wait_asynccnt)
    __builtin_amdgcn_s_wait_asynccnt(0);
#else
    asm volatile("s_wait_asynccnt 0x0" ::: "memory");
#endif
}

// ---------------------------------------------------------------------------
// f32 -> bf16 bulk convert (activations, once)
// ---------------------------------------------------------------------------
__global__ void convert_f32_bf16(const float* __restrict__ src,
                                 bf16_t* __restrict__ dst, int n) {
    int i = (blockIdx.x * blockDim.x + threadIdx.x) * 4;
    if (i >= n) return;
    v4f v = *(const v4f*)(src + i);
    v4bf o;
    o[0] = (bf16_t)v[0]; o[1] = (bf16_t)v[1];
    o[2] = (bf16_t)v[2]; o[3] = (bf16_t)v[3];
    *(v4bf*)(dst + i) = o;
}

// ---------------------------------------------------------------------------
// Build a 16x16x32 bf16 A/B fragment from LDS (row/col contiguous, stride 32).
// Wave32 layout: lanes 0-15 hold K 0-7 & 16-23, lanes 16-31 hold K 8-15 &
// 24-31 of row/col (lane&15). Two 16B ds loads per fragment.
// ---------------------------------------------------------------------------
__device__ inline v16bf load_frag(const bf16_t* base, int khalf) {
    const bf16_t* p = base + khalf * 8;
    v8bf lo = *(const v8bf*)(p);
    v8bf hi = *(const v8bf*)(p + 16);
    return __builtin_shufflevector(lo, hi, 0, 1, 2, 3, 4, 5, 6, 7,
                                           8, 9, 10, 11, 12, 13, 14, 15);
}

// ---------------------------------------------------------------------------
// GEMM1: gate_up = A(T x H, bf16) * W(H x 2D, f32->bf16) + bias, fused
// clamped-SiLU gating over (even, odd) column pairs -> gated (T x D, bf16).
// Block tile 128x128, BK=32, double-buffered LDS, 8 waves: 2(M) x 4(N).
// ---------------------------------------------------------------------------
__launch_bounds__(256)
__global__ void moe_gemm1(const bf16_t* __restrict__ A,     // T x H
                          const float*  __restrict__ W,     // H x GU (expert)
                          const float*  __restrict__ bias,  // GU
                          bf16_t*       __restrict__ gated) // T x D
{
    constexpr int BM = 128, BN = 128, BK = 32;
    __shared__ bf16_t As[2][BM * BK];
    __shared__ bf16_t Bs[2][BN * BK];   // transposed: [n][k]

    const int tid  = threadIdx.x;
    const int lane = tid & 31;
    const int wave = tid >> 5;
    const int wm   = wave >> 2;      // 0..1 -> 64 rows
    const int wn   = wave & 3;       // 0..3 -> 32 cols
    const int r0   = blockIdx.y * BM;
    const int c0   = blockIdx.x * BN;

    v8f acc[4][2] = {};

    const int lrow  = tid >> 1;               // A-tile row this thread stages
    const int lcol  = (tid & 1) * 16;         // 16 bf16 = 32B per thread
    const int bn4   = (tid & 31) * 4;         // B-tile col group
    const int kp    = (tid >> 5) * 2;         // B-tile k pair base (0..14)
    const int khalf = lane >> 4;

    auto stage = [&](int buf, int k0) {
        // A tile: async global->LDS (bf16, no conversion needed)
        async_copy32(&As[buf][lrow * BK + lcol],
                     A + (size_t)(r0 + lrow) * HDIM + k0 + lcol);
        // B tile: f32 global -> packed bf16 pairs, transposed into LDS
#pragma unroll
        for (int kk = 0; kk < 2; ++kk) {
            const int k = kp + kk * 16;
            v4f w0 = *(const v4f*)(W + (size_t)(k0 + k) * GUDIM + c0 + bn4);
            v4f w1 = *(const v4f*)(W + (size_t)(k0 + k + 1) * GUDIM + c0 + bn4);
#pragma unroll
            for (int q = 0; q < 4; ++q) {
                v2bf p;
                p[0] = (bf16_t)w0[q];
                p[1] = (bf16_t)w1[q];
                *(v2bf*)&Bs[buf][(bn4 + q) * BK + k] = p;
            }
        }
        if (k0 + BK < HDIM)
            __builtin_prefetch(W + (size_t)(k0 + BK + kp) * GUDIM + c0 + bn4, 0, 0);
    };

    auto compute = [&](int buf) {
        v16bf bfr[2];
#pragma unroll
        for (int j = 0; j < 2; ++j)
            bfr[j] = load_frag(&Bs[buf][(wn * 32 + j * 16 + (lane & 15)) * BK], khalf);
#pragma unroll
        for (int i = 0; i < 4; ++i) {
            v16bf afr = load_frag(&As[buf][(wm * 64 + i * 16 + (lane & 15)) * BK], khalf);
#pragma unroll
            for (int j = 0; j < 2; ++j)
                acc[i][j] = __builtin_amdgcn_wmma_f32_16x16x32_bf16(
                    false, afr, false, bfr[j], (short)0, acc[i][j], false, false);
        }
    };

    stage(0, 0);
    wait_async0();
    __syncthreads();
    for (int k0 = 0; k0 < HDIM; k0 += BK) {
        const int cur = (k0 / BK) & 1;
        if (k0 + BK < HDIM) stage(cur ^ 1, k0 + BK);
        compute(cur);
        wait_async0();
        __syncthreads();
    }

    // Epilogue: bias + pair (even=gate, odd=up) via lane-xor, write gated bf16
    const int rb = wm * 64 + (lane >> 4) * 8;
    const int cl = lane & 15;
#pragma unroll
    for (int j = 0; j < 2; ++j) {
        const int col = c0 + wn * 32 + j * 16 + cl;
        const float bv = bias[col];
#pragma unroll
        for (int i = 0; i < 4; ++i) {
#pragma unroll
            for (int r = 0; r < 8; ++r) {
                float self  = acc[i][j][r] + bv;
                float other = __shfl_xor(self, 1, 32);
                if ((lane & 1) == 0) {
                    float g   = fminf(self, ACT_LIMIT);
                    float u   = fminf(fmaxf(other, -ACT_LIMIT), ACT_LIMIT);
                    float glu = g / (1.0f + __expf(-g * ACT_ALPHA));
                    float gv  = (u + 1.0f) * glu;
                    const int t = r0 + rb + i * 16 + r;
                    gated[(size_t)t * DDIM + (col >> 1)] = (bf16_t)gv;
                }
            }
        }
    }
}

// ---------------------------------------------------------------------------
// GEMM2: out (+)= routing[t,e] * (gated(T x D, bf16) * Wd(D x H) + bias)
// Block tile 128x64 (2880 = 64*45), BK=32, double-buffered, 8 waves 4(M)x2(N).
// ---------------------------------------------------------------------------
__launch_bounds__(256)
__global__ void moe_gemm2(const bf16_t* __restrict__ Act,     // T x D
                          const float*  __restrict__ Wd,      // D x H (expert)
                          const float*  __restrict__ bias,    // H
                          const float*  __restrict__ routing, // T x E
                          int e, int first,
                          float* __restrict__ out)            // T x H
{
    constexpr int BM = 128, BN = 64, BK = 32;
    __shared__ bf16_t As[2][BM * BK];
    __shared__ bf16_t Bs[2][BN * BK];   // transposed: [n][k]

    const int tid  = threadIdx.x;
    const int lane = tid & 31;
    const int wave = tid >> 5;
    const int wm   = wave >> 1;      // 0..3 -> 32 rows
    const int wn   = wave & 1;       // 0..1 -> 32 cols
    const int r0   = blockIdx.y * BM;
    const int c0   = blockIdx.x * BN;

    v8f acc[2][2] = {};

    const int lrow  = tid >> 1;
    const int lcol  = (tid & 1) * 16;
    const int bn4   = (tid & 15) * 4;  // 0..60
    const int kp    = (tid >> 4) * 2;  // 0..30
    const int khalf = lane >> 4;

    auto stage = [&](int buf, int k0) {
        async_copy32(&As[buf][lrow * BK + lcol],
                     Act + (size_t)(r0 + lrow) * DDIM + k0 + lcol);
        v4f w0 = *(const v4f*)(Wd + (size_t)(k0 + kp) * HDIM + c0 + bn4);
        v4f w1 = *(const v4f*)(Wd + (size_t)(k0 + kp + 1) * HDIM + c0 + bn4);
#pragma unroll
        for (int q = 0; q < 4; ++q) {
            v2bf p;
            p[0] = (bf16_t)w0[q];
            p[1] = (bf16_t)w1[q];
            *(v2bf*)&Bs[buf][(bn4 + q) * BK + kp] = p;
        }
        if (k0 + BK < DDIM)
            __builtin_prefetch(Wd + (size_t)(k0 + BK + kp) * HDIM + c0 + bn4, 0, 0);
    };

    auto compute = [&](int buf) {
        v16bf bfr[2];
#pragma unroll
        for (int j = 0; j < 2; ++j)
            bfr[j] = load_frag(&Bs[buf][(wn * 32 + j * 16 + (lane & 15)) * BK], khalf);
#pragma unroll
        for (int i = 0; i < 2; ++i) {
            v16bf afr = load_frag(&As[buf][(wm * 32 + i * 16 + (lane & 15)) * BK], khalf);
#pragma unroll
            for (int j = 0; j < 2; ++j)
                acc[i][j] = __builtin_amdgcn_wmma_f32_16x16x32_bf16(
                    false, afr, false, bfr[j], (short)0, acc[i][j], false, false);
        }
    };

    stage(0, 0);
    wait_async0();
    __syncthreads();
    for (int k0 = 0; k0 < DDIM; k0 += BK) {
        const int cur = (k0 / BK) & 1;
        if (k0 + BK < DDIM) stage(cur ^ 1, k0 + BK);
        compute(cur);
        wait_async0();
        __syncthreads();
    }

    const int rb = wm * 32 + (lane >> 4) * 8;
    const int cl = lane & 15;
#pragma unroll
    for (int j = 0; j < 2; ++j) {
        const int col = c0 + wn * 32 + j * 16 + cl;
        const float bv = bias[col];
#pragma unroll
        for (int i = 0; i < 2; ++i) {
#pragma unroll
            for (int r = 0; r < 8; ++r) {
                const int t = r0 + rb + i * 16 + r;
                const float rw = routing[(size_t)t * NEXP + e];
                const float v  = rw * (acc[i][j][r] + bv);
                const size_t idx = (size_t)t * HDIM + col;
                if (first) out[idx] = v;
                else       out[idx] += v;
            }
        }
    }
}

// ---------------------------------------------------------------------------
extern "C" void kernel_launch(void* const* d_in, const int* in_sizes, int n_in,
                              void* d_out, int out_size, void* d_ws, size_t ws_size,
                              hipStream_t stream) {
    const float* hidden  = (const float*)d_in[0]; // (2,1024,2880)
    const float* routing = (const float*)d_in[1]; // (2048, 8)
    const float* gup     = (const float*)d_in[2]; // (8, 2880, 5760)
    const float* gup_b   = (const float*)d_in[3]; // (8, 5760)
    const float* dwn     = (const float*)d_in[4]; // (8, 2880, 2880)
    const float* dwn_b   = (const float*)d_in[5]; // (8, 2880)
    float* out = (float*)d_out;                   // (2048, 2880)

    bf16_t* Abf   = (bf16_t*)d_ws;                 // T*H bf16  (~11.8 MB)
    bf16_t* gated = Abf + (size_t)T_TOK * HDIM;    // T*D bf16  (~11.8 MB)

    {
        const int n = T_TOK * HDIM;
        convert_f32_bf16<<<n / (4 * 256), 256, 0, stream>>>(hidden, Abf, n);
    }

    dim3 g1(GUDIM / 128, T_TOK / 128); // (45, 16)
    dim3 g2(HDIM / 64,   T_TOK / 128); // (45, 16)
    for (int e = 0; e < NEXP; ++e) {
        moe_gemm1<<<g1, 256, 0, stream>>>(
            Abf, gup + (size_t)e * HDIM * GUDIM, gup_b + (size_t)e * GUDIM, gated);
        moe_gemm2<<<g2, 256, 0, stream>>>(
            gated, dwn + (size_t)e * DDIM * HDIM, dwn_b + (size_t)e * HDIM,
            routing, e, e == 0 ? 1 : 0, out);
    }
}
